// InfiniAttention_67645734913025
// MI455X (gfx1250) — matrix operand order repaired and checked by reading, not compiled, not run on hardware
//
#include <hip/hip_runtime.h>
#include <hip/hip_bf16.h>
#include <stddef.h>
#include <stdint.h>

// ---------------------------------------------------------------------------
// InfiniAttention on MI455X (gfx1250, wave32, WMMA + async global->LDS DMA).
// All GEMM operands are laid out row-major A[M,K] / Bt[N,K] in bf16 so LDS
// tiles are contiguous 64B runs -> staged with GLOBAL_LOAD_ASYNC_TO_LDS_B128
// (ASYNCcnt-tracked, double-buffered). Accumulation fp32 via
// v_wmma_f32_16x16x32_bf16.
// ---------------------------------------------------------------------------

typedef __attribute__((ext_vector_type(16))) __bf16 v16bf;
typedef __attribute__((ext_vector_type(8)))  float  v8f;

#define B_DIM 4
#define S_DIM 2048
#define E_DIM 1024
#define H_DIM 4
#define D_DIM 256

static __device__ __forceinline__ unsigned pack_bf16(float a, float b) {
  union { __bf16 h; unsigned short s; } ua, ub;
  ua.h = (__bf16)a; ub.h = (__bf16)b;
  return (unsigned)ua.s | ((unsigned)ub.s << 16);
}

// LDS byte offset of a __shared__ object (generic LDS address: low 32 bits).
static __device__ __forceinline__ unsigned lds_off_of(const void* p) {
  return (unsigned)(uintptr_t)p;
}

// Async DMA: 16 bytes global -> LDS, tracked by ASYNCcnt (CDNA5 ISA).
static __device__ __forceinline__ void async_ld_b128(unsigned lds_byte_addr,
                                                     const void* gptr) {
  asm volatile("global_load_async_to_lds_b128 %0, %1, off"
               :: "v"(lds_byte_addr), "v"((unsigned long long)(uintptr_t)gptr)
               : "memory");
}
static __device__ __forceinline__ void wait_asynccnt0() {
  asm volatile("s_wait_asynccnt 0x0" ::: "memory");
}

// ---------------------------------------------------------------------------
// bf16 WMMA GEMM: C[M,N] = A[M,K] * Bt[N,K]^T (+Cinit)(+bias row).
// Block tile 128x128, K-step 32, 256 threads = 8 waves (2 x 4), each wave a
// 64x32 patch = 4x2 WMMA tiles. Double-buffered async global->LDS staging.
// Requires M%128==0, N%128==0, K%32==0 (true for all launches below).
// ---------------------------------------------------------------------------
__global__ __launch_bounds__(256)
void gemm_bf16_wmma(const __bf16* __restrict__ A, const __bf16* __restrict__ Bt,
                    float* __restrict__ C, const float* __restrict__ Cinit, int ldci,
                    const float* __restrict__ bias,
                    int M, int N, int K, int lda, int ldb, int ldc)
{
  __shared__ __bf16 shA[2][128][32];
  __shared__ __bf16 shB[2][128][32];

  const int tid  = threadIdx.x;
  const int lane = tid & 31;
  const int wid  = tid >> 5;
  const int wr   = wid >> 2;        // 0..1 : 64-row band
  const int wc   = wid & 3;         // 0..3 : 32-col band
  const int half = lane >> 4;
  const int ml   = lane & 15;
  const int rowB = blockIdx.y * 128;
  const int colB = blockIdx.x * 128;

  // Per-thread DMA chunk assignment: 512 16B-chunks per tile, 2 per thread.
  // chunk c -> row r = c>>2, segment s = (c&3)*8 elements.
  const int r0 = tid >> 2,         s0 = (tid & 3) * 8;
  const int r1 = (tid + 256) >> 2, s1 = ((tid + 256) & 3) * 8;

  const __bf16* gA0 = A  + (size_t)(rowB + r0) * lda + s0;
  const __bf16* gA1 = A  + (size_t)(rowB + r1) * lda + s1;
  const __bf16* gB0 = Bt + (size_t)(colB + r0) * ldb + s0;
  const __bf16* gB1 = Bt + (size_t)(colB + r1) * ldb + s1;

  const unsigned ldsA = lds_off_of(&shA[0][0][0]);
  const unsigned ldsB = lds_off_of(&shB[0][0][0]);
  const unsigned oA0 = (unsigned)((r0 * 32 + s0) * 2);
  const unsigned oA1 = (unsigned)((r1 * 32 + s1) * 2);

  auto issue = [&](int buf, int k0) {
    const unsigned bo = (unsigned)buf * (128u * 32u * 2u);
    async_ld_b128(ldsA + bo + oA0, gA0 + k0);
    async_ld_b128(ldsA + bo + oA1, gA1 + k0);
    async_ld_b128(ldsB + bo + oA0, gB0 + k0);
    async_ld_b128(ldsB + bo + oA1, gB1 + k0);
  };

  const v8f vzero = {0.f,0.f,0.f,0.f,0.f,0.f,0.f,0.f};
  v8f acc[4][2];
  for (int tm = 0; tm < 4; ++tm)
    for (int tn = 0; tn < 2; ++tn) acc[tm][tn] = vzero;

  issue(0, 0);
  int buf = 0;
  for (int k0 = 0; k0 < K; k0 += 32) {
    wait_asynccnt0();
    __syncthreads();                      // tile `buf` resident for all waves
    if (k0 + 32 < K) issue(buf ^ 1, k0 + 32);

    union F { v16bf v; unsigned u[8]; } af[4], bfr[2];
    for (int tm = 0; tm < 4; ++tm) {
      const int mrow = wr * 64 + tm * 16 + ml;
      for (int j = 0; j < 8; ++j) {
        const int ka = half * 16 + j * 2;                          // A 16x32 layout (contiguous k)
        af[tm].u[j] = *(const unsigned*)&shA[buf][mrow][ka];
      }
    }
    for (int tn = 0; tn < 2; ++tn) {
      const int ncol = wc * 32 + tn * 16 + ml;
      for (int j = 0; j < 8; ++j) {
        const int kb = half * 16 + j * 2;                          // B 32x16 layout
        bfr[tn].u[j] = *(const unsigned*)&shB[buf][ncol][kb];
      }
    }
    for (int tm = 0; tm < 4; ++tm)
      for (int tn = 0; tn < 2; ++tn)
        acc[tm][tn] = __builtin_amdgcn_wmma_f32_16x16x32_bf16(
            false, af[tm].v, false, bfr[tn].v, (short)0, acc[tm][tn], false, false);
    __syncthreads();                      // everyone done reading `buf`
    buf ^= 1;
  }

  for (int tm = 0; tm < 4; ++tm)
    for (int tn = 0; tn < 2; ++tn)
      for (int r = 0; r < 8; ++r) {
        const int m = rowB + wr * 64 + tm * 16 + r + half * 8;
        const int n = colB + wc * 32 + tn * 16 + ml;
        float base = 0.f;
        if (Cinit) base += Cinit[(size_t)m * ldci + n];
        if (bias)  base += bias[n];
        C[(size_t)m * ldc + n] = acc[tm][tn][r] + base;
      }
}

// ---------------------------------------------------------------------------
// Flash-style causal attention (no scaling, per reference). One wave owns a
// 16-query tile; Q resident as 8 bf16 A-fragments; 32-key K/V blocks staged
// to LDS; fp32 online softmax; P*V via WMMA into a 16x256 fp32 accumulator.
// Grid: (S/128, H, B), 256 threads.
// ---------------------------------------------------------------------------
__global__ __launch_bounds__(256)
void flash_attn_causal(const float* __restrict__ q, const float* __restrict__ k,
                       const float* __restrict__ v, float* __restrict__ attnL)
{
  __shared__ __bf16 shK[32][256];       // [key][d]
  __shared__ __bf16 shV[256][32];       // [d][key]
  __shared__ __bf16 shP[8][16][32];     // per-wave P staging

  const int tid  = threadIdx.x;
  const int lane = tid & 31;
  const int w    = tid >> 5;
  const int half = lane >> 4;
  const int ml   = lane & 15;
  const int qbase = blockIdx.x * 128;
  const int head  = blockIdx.y;
  const int b     = blockIdx.z;
  const int qw0   = qbase + w * 16;

  union F { v16bf v; unsigned u[8]; };

  F qf[8];
  {
    const float* qp = q + (size_t)(b * S_DIM + qw0 + ml) * E_DIM + head * D_DIM;
    for (int c = 0; c < 8; ++c)
      for (int j = 0; j < 8; ++j) {
        const int d = c * 32 + half * 16 + j * 2;
        qf[c].u[j] = pack_bf16(qp[d], qp[d + 1]);
      }
  }

  const v8f vzero = {0.f,0.f,0.f,0.f,0.f,0.f,0.f,0.f};
  v8f acc[16];
  for (int t = 0; t < 16; ++t) acc[t] = vzero;
  float mrow[8], lrow[8];
  for (int r = 0; r < 8; ++r) { mrow[r] = -3.0e38f; lrow[r] = 0.f; }

  const int nkb = qbase / 32 + 4;
  for (int kb = 0; kb < nkb; ++kb) {
    const int k0 = kb * 32;
    for (int i = 0; i < 32; ++i) {
      const int e = tid + i * 256;
      const int key = e >> 8, d = e & 255;
      const size_t g = (size_t)(b * S_DIM + k0 + key) * E_DIM + head * D_DIM + d;
      shK[key][d] = (__bf16)k[g];
      shV[d][key] = (__bf16)v[g];
    }
    if (kb + 1 < nkb)
      __builtin_prefetch(&k[(size_t)(b * S_DIM + k0 + 32) * E_DIM + head * D_DIM], 0, 1);
    __syncthreads();

    if (k0 <= qw0 + 15) {
      v8f sc0 = vzero, sc1 = vzero;
      for (int c = 0; c < 8; ++c) {
        F bk0, bk1;
        for (int j = 0; j < 8; ++j) {
          const int d = c * 32 + half * 16 + j * 2;
          bk0.u[j] = *(const unsigned*)&shK[ml][d];
          bk1.u[j] = *(const unsigned*)&shK[16 + ml][d];
        }
        sc0 = __builtin_amdgcn_wmma_f32_16x16x32_bf16(false, qf[c].v, false, bk0.v, (short)0, sc0, false, false);
        sc1 = __builtin_amdgcn_wmma_f32_16x16x32_bf16(false, qf[c].v, false, bk1.v, (short)0, sc1, false, false);
      }
      for (int r = 0; r < 8; ++r) {
        const int qrow = qw0 + r + 8 * half;
        float x0 = (k0 + ml      <= qrow) ? sc0[r] : -3.0e38f;
        float x1 = (k0 + 16 + ml <= qrow) ? sc1[r] : -3.0e38f;
        float mx = fmaxf(x0, x1);
        for (int off = 1; off < 16; off <<= 1) mx = fmaxf(mx, __shfl_xor(mx, off, 16));
        const float mnew  = fmaxf(mrow[r], mx);
        const float scale = __expf(mrow[r] - mnew);
        const float p0 = __expf(x0 - mnew);
        const float p1 = __expf(x1 - mnew);
        float ps = p0 + p1;
        for (int off = 1; off < 16; off <<= 1) ps += __shfl_xor(ps, off, 16);
        lrow[r] = lrow[r] * scale + ps;
        mrow[r] = mnew;
        for (int t = 0; t < 16; ++t) acc[t][r] *= scale;
        shP[w][r + 8 * half][ml]      = (__bf16)p0;
        shP[w][r + 8 * half][16 + ml] = (__bf16)p1;
      }
      asm volatile("s_wait_dscnt 0" ::: "memory");   // wave-local LDS RAW
      F pf;
      for (int j = 0; j < 8; ++j) {
        const int kk = half * 16 + j * 2;
        pf.u[j] = *(const unsigned*)&shP[w][ml][kk];
      }
      for (int t = 0; t < 16; ++t) {
        F bv;
        for (int j = 0; j < 8; ++j) {
          const int key = half * 16 + j * 2;
          bv.u[j] = *(const unsigned*)&shV[t * 16 + ml][key];
        }
        acc[t] = __builtin_amdgcn_wmma_f32_16x16x32_bf16(false, pf.v, false, bv.v, (short)0, acc[t], false, false);
      }
    }
    __syncthreads();
  }

  for (int t = 0; t < 16; ++t)
    for (int r = 0; r < 8; ++r) {
      const int qrow = qw0 + r + 8 * half;
      attnL[(size_t)(b * S_DIM + qrow) * E_DIM + head * D_DIM + t * 16 + ml] =
          acc[t][r] / lrow[r];
    }
}

// ---------------------------------------------------------------------------
// Elementwise glue.
// ---------------------------------------------------------------------------
__global__ void cvt_f32_bf16(const float* __restrict__ in, __bf16* __restrict__ out, int n) {
  const int i = blockIdx.x * blockDim.x + threadIdx.x;
  if (i < n) out[i] = (__bf16)in[i];
}

// Batched transpose-convert: out[b][c][r] = in[b][r][c]  (R x C per batch).
__global__ void cvt_t_f32_bf16(const float* __restrict__ in, __bf16* __restrict__ out,
                               int R, int Cc, int nb) {
  const int i = blockIdx.x * blockDim.x + threadIdx.x;
  if (i >= R * Cc * nb) return;
  const int bidx = i / (R * Cc);
  const int rem  = i - bidx * R * Cc;
  const int r = rem / Cc, c = rem - r * Cc;
  out[(size_t)bidx * R * Cc + (size_t)c * R + r] = (__bf16)in[i];
}

// elu(x)+1 == (x>0 ? x+1 : exp(x)); writes sigma_q, sigma_k, sigma_k^T.
__global__ void featmap_kernel(const float* __restrict__ q, const float* __restrict__ k,
                               __bf16* __restrict__ sq, __bf16* __restrict__ sk,
                               __bf16* __restrict__ skT, int n) {
  const int i = blockIdx.x * blockDim.x + threadIdx.x;
  if (i >= n) return;
  const int e = i % E_DIM;
  const int s = (i / E_DIM) % S_DIM;
  const int b = i / (E_DIM * S_DIM);
  const int h = e / D_DIM, d = e % D_DIM;
  const float x = q[i];
  sq[i] = (__bf16)((x > 0.f) ? (x + 1.f) : __expf(x));
  const float y = k[i];
  const __bf16 sgk = (__bf16)((y > 0.f) ? (y + 1.f) : __expf(y));
  sk[i] = sgk;
  skT[((size_t)(b * H_DIM + h) * D_DIM + d) * S_DIM + s] = sgk;
}

__global__ void rowdot_z_kernel(const __bf16* __restrict__ sig, const float* __restrict__ z,
                                float* __restrict__ den) {
  const int i = blockIdx.x * blockDim.x + threadIdx.x;
  if (i >= B_DIM * H_DIM * S_DIM) return;
  const int s = i % S_DIM;
  const int h = (i / S_DIM) % H_DIM;
  const int b = i / (S_DIM * H_DIM);
  const __bf16* p = sig + (size_t)(b * S_DIM + s) * E_DIM + h * D_DIM;
  const float* zp = z + h * D_DIM;
  float acc = 0.f;
  for (int d = 0; d < D_DIM; ++d) acc += (float)p[d] * zp[d];
  den[i] = acc;
}

__global__ void combine_kernel(const float* __restrict__ Anum, const float* __restrict__ attnL,
                               const float* __restrict__ denQ, const float* __restrict__ betas,
                               __bf16* __restrict__ comb, int n) {
  const int i = blockIdx.x * blockDim.x + threadIdx.x;
  if (i >= n) return;
  const int e = i % E_DIM;
  const int s = (i / E_DIM) % S_DIM;
  const int b = i / (E_DIM * S_DIM);
  const int h = e / D_DIM;
  const float g   = 1.f / (1.f + __expf(-betas[h]));
  const float den = denQ[(b * H_DIM + h) * S_DIM + s] + 1e-6f;
  comb[i] = (__bf16)(g * (Anum[i] / den) + (1.f - g) * attnL[i]);
}

// uT[b,h][e'][s] = bf16(v - delta)   (transposed for the update GEMM's Bt).
__global__ void residual_kernel(const float* __restrict__ v, const float* __restrict__ Dnum,
                                const float* __restrict__ denK, __bf16* __restrict__ uT, int n) {
  const int i = blockIdx.x * blockDim.x + threadIdx.x;
  if (i >= n) return;
  const int e = i % E_DIM;
  const int s = (i / E_DIM) % S_DIM;
  const int b = i / (E_DIM * S_DIM);
  const int h = e / D_DIM, d = e % D_DIM;
  const float den = denK[(b * H_DIM + h) * S_DIM + s] + 1e-6f;
  uT[((size_t)(b * H_DIM + h) * D_DIM + d) * S_DIM + s] = (__bf16)(v[i] - Dnum[i] / den);
}

__global__ void znew_kernel(const __bf16* __restrict__ skT, const float* __restrict__ z,
                            float* __restrict__ outz) {
  const int i = blockIdx.x * blockDim.x + threadIdx.x;
  if (i >= B_DIM * H_DIM * D_DIM) return;
  const __bf16* p = skT + (size_t)i * S_DIM;   // contiguous per (b,h,d)
  float acc = 0.f;
  for (int s = 0; s < S_DIM; ++s) acc += (float)p[s];
  outz[i] = z[i % (H_DIM * D_DIM)] + acc;      // z broadcast over batch: [h][d]
}

// ---------------------------------------------------------------------------
// Host orchestration.
// ---------------------------------------------------------------------------
extern "C" void kernel_launch(void* const* d_in, const int* in_sizes, int n_in,
                              void* d_out, int out_size, void* d_ws, size_t ws_size,
                              hipStream_t stream) {
  (void)in_sizes; (void)n_in; (void)out_size; (void)ws_size;
  const float* hs    = (const float*)d_in[0];
  const float* Wq    = (const float*)d_in[1];
  const float* Wk    = (const float*)d_in[2];
  const float* Wv    = (const float*)d_in[3];
  const float* Wo    = (const float*)d_in[4];
  const float* bo    = (const float*)d_in[5];
  const float* betas = (const float*)d_in[6];
  const float* mem   = (const float*)d_in[7];
  const float* z     = (const float*)d_in[8];

  const size_t BSE = (size_t)B_DIM * S_DIM * E_DIM;
  const int   nBSE = (int)BSE;
  char* ws = (char*)d_ws;
  size_t off = 0;
  auto take = [&](size_t bytes) -> char* {
    char* p = ws + off;
    off += (bytes + 255) & ~(size_t)255;
    return p;
  };

  __bf16* hsb   = (__bf16*)take(BSE * 2);                      // reused as `comb`
  __bf16* WqT   = (__bf16*)take((size_t)E_DIM * E_DIM * 2);
  __bf16* WkT   = (__bf16*)take((size_t)E_DIM * E_DIM * 2);
  __bf16* WvT   = (__bf16*)take((size_t)E_DIM * E_DIM * 2);
  __bf16* WoT   = (__bf16*)take((size_t)E_DIM * E_DIM * 2);
  __bf16* memT  = (__bf16*)take((size_t)H_DIM * D_DIM * D_DIM * 2);
  float*  qf    = (float*)take(BSE * 4);
  float*  kf    = (float*)take(BSE * 4);
  float*  vf    = (float*)take(BSE * 4);
  __bf16* sqb   = (__bf16*)take(BSE * 2);
  __bf16* skb   = (__bf16*)take(BSE * 2);
  __bf16* skT   = (__bf16*)take(BSE * 2);
  float*  Anum  = (float*)take(BSE * 4);
  float*  Dnum  = (float*)take(BSE * 4);
  float*  attnL = (float*)take(BSE * 4);
  __bf16* uT    = (__bf16*)take(BSE * 2);
  float*  denQ  = (float*)take((size_t)B_DIM * H_DIM * S_DIM * 4);
  float*  denK  = (float*)take((size_t)B_DIM * H_DIM * S_DIM * 4);
  __bf16* comb  = hsb;

  float* out_main = (float*)d_out;
  float* out_mem  = out_main + BSE;
  float* out_z    = out_mem + (size_t)B_DIM * H_DIM * D_DIM * D_DIM;

  const dim3 blk(256);
  const int EE = E_DIM * E_DIM;
  const int nMem = H_DIM * D_DIM * D_DIM;

  // 1. Conversions (weights/mem transposed to Bt layout).
  cvt_f32_bf16<<<dim3((nBSE + 255) / 256), blk, 0, stream>>>(hs, hsb, nBSE);
  cvt_t_f32_bf16<<<dim3((EE + 255) / 256), blk, 0, stream>>>(Wq, WqT, E_DIM, E_DIM, 1);
  cvt_t_f32_bf16<<<dim3((EE + 255) / 256), blk, 0, stream>>>(Wk, WkT, E_DIM, E_DIM, 1);
  cvt_t_f32_bf16<<<dim3((EE + 255) / 256), blk, 0, stream>>>(Wv, WvT, E_DIM, E_DIM, 1);
  cvt_t_f32_bf16<<<dim3((EE + 255) / 256), blk, 0, stream>>>(Wo, WoT, E_DIM, E_DIM, 1);
  cvt_t_f32_bf16<<<dim3((nMem + 255) / 256), blk, 0, stream>>>(mem, memT, D_DIM, D_DIM, H_DIM);

  // 2. QKV projections [8192,1024,1024].
  const dim3 gProj(E_DIM / 128, (B_DIM * S_DIM) / 128);
  gemm_bf16_wmma<<<gProj, blk, 0, stream>>>(hsb, WqT, qf, nullptr, 0, nullptr,
      B_DIM * S_DIM, E_DIM, E_DIM, E_DIM, E_DIM, E_DIM);
  gemm_bf16_wmma<<<gProj, blk, 0, stream>>>(hsb, WkT, kf, nullptr, 0, nullptr,
      B_DIM * S_DIM, E_DIM, E_DIM, E_DIM, E_DIM, E_DIM);
  gemm_bf16_wmma<<<gProj, blk, 0, stream>>>(hsb, WvT, vf, nullptr, 0, nullptr,
      B_DIM * S_DIM, E_DIM, E_DIM, E_DIM, E_DIM, E_DIM);

  // 3. Feature maps (+ sigma_k^T for the update GEMM).
  featmap_kernel<<<dim3((nBSE + 255) / 256), blk, 0, stream>>>(qf, kf, sqb, skb, skT, nBSE);

  // 4. Denominators sigma . z.
  const int nBHS = B_DIM * H_DIM * S_DIM;
  rowdot_z_kernel<<<dim3((nBHS + 255) / 256), blk, 0, stream>>>(sqb, z, denQ);
  rowdot_z_kernel<<<dim3((nBHS + 255) / 256), blk, 0, stream>>>(skb, z, denK);

  // 5. Memory-read numerators per head: [8192,256,256].
  const dim3 gHead(D_DIM / 128, (B_DIM * S_DIM) / 128);
  for (int h = 0; h < H_DIM; ++h) {
    gemm_bf16_wmma<<<gHead, blk, 0, stream>>>(sqb + h * D_DIM, memT + (size_t)h * D_DIM * D_DIM,
        Anum + h * D_DIM, nullptr, 0, nullptr,
        B_DIM * S_DIM, D_DIM, D_DIM, E_DIM, D_DIM, E_DIM);
    gemm_bf16_wmma<<<gHead, blk, 0, stream>>>(skb + h * D_DIM, memT + (size_t)h * D_DIM * D_DIM,
        Dnum + h * D_DIM, nullptr, 0, nullptr,
        B_DIM * S_DIM, D_DIM, D_DIM, E_DIM, D_DIM, E_DIM);
  }

  // 6. Causal flash attention.
  flash_attn_causal<<<dim3(S_DIM / 128, H_DIM, B_DIM), blk, 0, stream>>>(qf, kf, vf, attnL);

  // 7. Gate mix -> bf16 for output projection.
  combine_kernel<<<dim3((nBSE + 255) / 256), blk, 0, stream>>>(Anum, attnL, denQ, betas, comb, nBSE);

  // 8. uT = (v - delta)^T (bf16).
  residual_kernel<<<dim3((nBSE + 255) / 256), blk, 0, stream>>>(vf, Dnum, denK, uT, nBSE);

  // 9. mem_new[b,h] = mem[h] + sigma_k^T @ u : [256,256,2048] GEMMs, A=skT, Bt=uT.
  const dim3 gUpd(D_DIM / 128, D_DIM / 128);
  for (int b = 0; b < B_DIM; ++b)
    for (int h = 0; h < H_DIM; ++h) {
      const size_t base = (size_t)(b * H_DIM + h) * D_DIM * S_DIM;
      gemm_bf16_wmma<<<gUpd, blk, 0, stream>>>(skT + base, uT + base,
          out_mem + (size_t)(b * H_DIM + h) * D_DIM * D_DIM,
          mem + (size_t)h * D_DIM * D_DIM, D_DIM, nullptr,
          D_DIM, D_DIM, S_DIM, S_DIM, S_DIM, D_DIM);
    }

  // 10. z_new (contiguous reduction over skT).
  const int nBHD = B_DIM * H_DIM * D_DIM;
  znew_kernel<<<dim3((nBHD + 255) / 256), blk, 0, stream>>>(skT, z, out_z);

  // 11. Output projection + bias.
  gemm_bf16_wmma<<<gProj, blk, 0, stream>>>(comb, WoT, out_main, nullptr, 0, bo,
      B_DIM * S_DIM, E_DIM, E_DIM, E_DIM, E_DIM, E_DIM);
}
